// PitchTransformerOutputProj_40793599377848
// MI455X (gfx1250) — matrix operand order repaired and checked by reading, not compiled, not run on hardware
//
#include <hip/hip_runtime.h>
#include <hip/hip_fp16.h>

typedef __attribute__((ext_vector_type(16))) _Float16 v16h;
typedef __attribute__((ext_vector_type(8)))  _Float16 v8h;
typedef __attribute__((ext_vector_type(8)))  float    v8f;

// ---- workspace layout (halves), per layer ----
// qkvT  : 384 x 128   @ 0
// projT : 128 x 128   @ 49152
// fc1T  : 512 x 128   @ 65536
// fc2T  : 128 x 512   @ 131072
#define LWS 196608
#define SMEM_BYTES 264704   // h 64K | y16 32K | q16 32K | k16 32K | vT 32K | S 64K | extras

struct KParams {
  const float* x; const float* target;
  const float* cond_w; const float* cond_b;
  const float* inp_w;  const float* inp_b;
  const float* sos;
  const float* ln1_g[2]; const float* ln1_b[2];
  const float* qkv_b[2];
  const float* qn_g[2];  const float* qn_b[2];
  const float* kn_g[2];  const float* kn_b[2];
  const float* proj_b[2];
  const float* ln2_g[2]; const float* ln2_b[2];
  const float* fc1_b[2]; const float* fc2_b[2];
  const float* norm_g; const float* norm_b;
  const float* head_w; const float* head_b;
  const _Float16* wh;   // f16 transposed weights in d_ws
  float* out;
};

// A fragment (16x32 f16): lane holds row m = lane&15; halves e<8 -> k=hi*8+e,
// e>=8 -> k=16+hi*8+(e-8). Two contiguous 16B loads per lane.
__device__ inline v16h load_a(const _Float16* X, int ld, int m0, int k0, int lane) {
  int mr = lane & 15, hi = (lane >> 4) & 1;
  const _Float16* p = X + (size_t)(m0 + mr) * ld + k0 + hi * 8;
  v8h lo = *(const v8h*)p;
  v8h hh = *(const v8h*)(p + 16);
  v16h a;
#pragma unroll
  for (int i = 0; i < 8; ++i) { a[i] = lo[i]; a[i + 8] = hh[i]; }
  return a;
}

// B fragment (32x16 f16) from a transposed (N x K row-major) buffer:
// lane holds column n = lane&15; halves e<8 -> k=khi*8+e, e>=8 -> k=16+khi*8+(e-8). Two 16B loads.
__device__ inline v16h load_bm(const _Float16* WT, int ld, int n0, int k0, int lane) {
  int n = lane & 15, khi = (lane >> 4) & 1;
  const _Float16* p = WT + (size_t)(n0 + n) * ld + k0 + khi * 8;
  v8h lo = *(const v8h*)p;
  v8h hh = *(const v8h*)(p + 16);
  v16h b;
#pragma unroll
  for (int i = 0; i < 8; ++i) { b[i] = lo[i]; b[i + 8] = hh[i]; }
  return b;
}

__device__ inline v8f wmma_f16(v16h a, v16h b, v8f c) {
  return __builtin_amdgcn_wmma_f32_16x16x32_f16(false, a, false, b, (short)0, c, false, false);
}

__device__ inline void ln_rows_f16(const float* src, _Float16* dst,
                                   const float* g, const float* b, int tid) {
  if (tid < 128) {
    const float* r = src + tid * 128;
    float m = 0.f;
    for (int j = 0; j < 128; ++j) m += r[j];
    m *= (1.f / 128.f);
    float v = 0.f;
    for (int j = 0; j < 128; ++j) { float d = r[j] - m; v += d * d; }
    float rs = rsqrtf(v * (1.f / 128.f) + 1e-5f);
    _Float16* o = dst + tid * 128;
    for (int j = 0; j < 128; ++j) o[j] = (_Float16)((r[j] - m) * rs * g[j] + b[j]);
  }
}

__global__ void w_to_f16_t(const float* __restrict__ W, _Float16* __restrict__ out,
                           int K, int N) {
  int i = blockIdx.x * blockDim.x + threadIdx.x;
  if (i < K * N) {
    int n = i / K, k = i - n * K;
    out[i] = (_Float16)W[(size_t)k * N + n];   // out[n*K+k] = W[k][n]
  }
}

__global__ __launch_bounds__(256)
void pitchformer_main(KParams kp) {
  extern __shared__ char smem[];
  float*    h    = (float*)smem;                     // 128x128 f32
  _Float16* bufY = (_Float16*)(smem + 65536);        // LN out (A operand)
  _Float16* bufQ = (_Float16*)(smem + 98304);        // q16 -> attn16 -> z16
  _Float16* bufK = (_Float16*)(smem + 131072);       // k16 -> o16
  _Float16* bufV = (_Float16*)(smem + 163840);       // vT [hd][pos]
  float*    S    = (float*)(smem + 196608);          // scores f32 / final LN f32
  float*    condv = (float*)(smem + 262144);         // 128
  float*    cpart = (float*)(smem + 262656);         // 2x128
  float*    inpl  = (float*)(smem + 263680);         // 128x2

  const int tid = threadIdx.x;
  const int wave = tid >> 5, lane = tid & 31;
  const int s = blockIdx.x, bb = s >> 8, tt = s & 255;

  // ---- cond = x[s] @ Wc + bc  (split-K over 2 half-threads) ----
  {
    int j = tid & 127, half = tid >> 7;
    const float* xr = kp.x + (size_t)s * 512 + half * 256;
    float acc = 0.f;
    for (int e = 0; e < 256; ++e) acc += xr[e] * kp.cond_w[(size_t)(half * 256 + e) * 128 + j];
    cpart[half * 128 + j] = acc;
  }
  // ---- inp_seq (shifted target + sos) ----
  {
    int p = tid >> 1, oc = tid & 1;
    float v;
    if (p == 0) v = kp.sos[oc];
    else        v = kp.target[((size_t)(bb * 2 + oc) * 256 + tt) * 128 + (p - 1)];
    inpl[p * 2 + oc] = v;
  }
  __syncthreads();
  if (tid < 128) condv[tid] = cpart[tid] + cpart[128 + tid] + kp.cond_b[tid];
  __syncthreads();
  // ---- h init ----
  for (int idx = tid; idx < 128 * 128; idx += 256) {
    int p = idx >> 7, j = idx & 127;
    h[idx] = inpl[p * 2] * kp.inp_w[j] + inpl[p * 2 + 1] * kp.inp_w[128 + j]
           + kp.inp_b[j] + condv[j];
  }
  __syncthreads();

  for (int l = 0; l < 2; ++l) {
    const _Float16* wbase = kp.wh + (size_t)l * LWS;
    const _Float16* qkvT  = wbase;
    const _Float16* projT = wbase + 49152;
    const _Float16* fc1T  = wbase + 65536;
    const _Float16* fc2T  = wbase + 131072;

    ln_rows_f16(h, bufY, kp.ln1_g[l], kp.ln1_b[l], tid);
    __syncthreads();

    // ---- qkv GEMM: (128x128) @ (128x384) ----
    for (int tile = wave; tile < 8 * 24; tile += 8) {
      int mt = tile & 7, nt = tile >> 3;
      v8f acc = {0,0,0,0,0,0,0,0};
#pragma unroll
      for (int kk = 0; kk < 4; ++kk) {
        v16h a = load_a(bufY, 128, mt * 16, kk * 32, lane);
        v16h b = load_bm(qkvT, 128, nt * 16, kk * 32, lane);
        acc = wmma_f16(a, b, acc);
      }
      int hi = (lane >> 4) & 1;
      int ncol = nt * 16 + (lane & 15);
      float bias = kp.qkv_b[l][ncol];
      if (nt < 16) {                       // q or k, normal [pos][hd] layout
        _Float16* dst = (nt < 8) ? bufQ : bufK;
        int cc = ncol & 127;
#pragma unroll
        for (int r = 0; r < 8; ++r) {
          int m = mt * 16 + hi * 8 + r;
          dst[m * 128 + cc] = (_Float16)(acc[r] + bias);
        }
      } else {                             // v stored transposed vT[hd][pos]
        int hd = ncol - 256;
        v8h pk;
#pragma unroll
        for (int r = 0; r < 8; ++r) pk[r] = (_Float16)(acc[r] + bias);
        *(v8h*)(bufV + hd * 128 + mt * 16 + hi * 8) = pk;
      }
    }
    __syncthreads();

    // ---- q/k LayerNorm + RoPE (q also scaled by 1/sqrt(HD)) ----
    {
      int who = tid >> 7, p = tid & 127;
      _Float16* row = (who ? bufK : bufQ) + p * 128;
      float m = 0.f;
      for (int j = 0; j < 128; ++j) m += (float)row[j];
      m *= (1.f / 128.f);
      float v = 0.f;
      for (int j = 0; j < 128; ++j) { float d = (float)row[j] - m; v += d * d; }
      float rs = rsqrtf(v * (1.f / 128.f) + 1e-5f);
      const float* g  = who ? kp.kn_g[l] : kp.qn_g[l];
      const float* be = who ? kp.kn_b[l] : kp.qn_b[l];
      float qs = who ? 1.f : 0.08838834764831845f;  // 1/sqrt(128)
      for (int d = 0; d < 64; ++d) {
        float x1 = ((float)row[d]      - m) * rs * g[d]      + be[d];
        float x2 = ((float)row[d + 64] - m) * rs * g[d + 64] + be[d + 64];
        float ang = (float)p * expf((float)d * -0.14391156509505908f); // 10000^(-d/64)
        float sn, cs;
        sincosf(ang, &sn, &cs);
        row[d]      = (_Float16)((x1 * cs - x2 * sn) * qs);
        row[d + 64] = (_Float16)((x2 * cs + x1 * sn) * qs);
      }
    }
    __syncthreads();

    // ---- scores = q @ k^T (+ causal mask) ----
    for (int tile = wave; tile < 64; tile += 8) {
      int mt = tile & 7, nt = tile >> 3;
      v8f acc = {0,0,0,0,0,0,0,0};
#pragma unroll
      for (int kk = 0; kk < 4; ++kk) {
        v16h a = load_a(bufQ, 128, mt * 16, kk * 32, lane);
        v16h b = load_bm(bufK, 128, nt * 16, kk * 32, lane);  // k16[n][k] == (k^T)[k][n]
        acc = wmma_f16(a, b, acc);
      }
      int hi = (lane >> 4) & 1, n = nt * 16 + (lane & 15);
#pragma unroll
      for (int r = 0; r < 8; ++r) {
        int m = mt * 16 + hi * 8 + r;
        S[m * 128 + n] = (n <= m) ? acc[r] : -3.0e38f;
      }
    }
    __syncthreads();

    // ---- softmax rows -> attn f16 (reuse bufQ) ----
    if (tid < 128) {
      float* r = S + tid * 128;
      float mx = r[0];
      for (int j = 1; j < 128; ++j) mx = fmaxf(mx, r[j]);
      float sum = 0.f;
      for (int j = 0; j < 128; ++j) { float e = expf(r[j] - mx); r[j] = e; sum += e; }
      float inv = 1.f / sum;
      for (int j = 0; j < 128; ++j) bufQ[tid * 128 + j] = (_Float16)(r[j] * inv);
    }
    __syncthreads();

    // ---- o = attn @ v  (B from vT) -> o16 (reuse bufK) ----
    for (int tile = wave; tile < 64; tile += 8) {
      int mt = tile & 7, nt = tile >> 3;
      v8f acc = {0,0,0,0,0,0,0,0};
#pragma unroll
      for (int kk = 0; kk < 4; ++kk) {
        v16h a = load_a(bufQ, 128, mt * 16, kk * 32, lane);
        v16h b = load_bm(bufV, 128, nt * 16, kk * 32, lane);  // vT[hd][pos]
        acc = wmma_f16(a, b, acc);
      }
      int hi = (lane >> 4) & 1, n = nt * 16 + (lane & 15);
#pragma unroll
      for (int r = 0; r < 8; ++r)
        bufK[(mt * 16 + hi * 8 + r) * 128 + n] = (_Float16)acc[r];
    }
    __syncthreads();

    // ---- h += o @ Wproj + b ----
    for (int tile = wave; tile < 64; tile += 8) {
      int mt = tile & 7, nt = tile >> 3;
      v8f acc = {0,0,0,0,0,0,0,0};
#pragma unroll
      for (int kk = 0; kk < 4; ++kk) {
        v16h a = load_a(bufK, 128, mt * 16, kk * 32, lane);
        v16h b = load_bm(projT, 128, nt * 16, kk * 32, lane);
        acc = wmma_f16(a, b, acc);
      }
      int hi = (lane >> 4) & 1, n = nt * 16 + (lane & 15);
      float bias = kp.proj_b[l][n];
#pragma unroll
      for (int r = 0; r < 8; ++r) {
        int m = mt * 16 + hi * 8 + r;
        h[m * 128 + n] += acc[r] + bias;
      }
    }
    __syncthreads();

    ln_rows_f16(h, bufY, kp.ln2_g[l], kp.ln2_b[l], tid);
    __syncthreads();

    // ---- FFN: h += gelu(y@fc1+b1)@fc2+b2, K of fc2 chunked by 128 ----
    {
      v8f acc2[8];
#pragma unroll
      for (int i = 0; i < 8; ++i) acc2[i] = (v8f){0,0,0,0,0,0,0,0};
      for (int c = 0; c < 4; ++c) {
        // z_c = gelu(y @ fc1[:, c*128 : c*128+128] + b1) -> bufQ (f16)
        for (int tile = wave; tile < 64; tile += 8) {
          int mt = tile & 7, nt = tile >> 3;          // mt == wave
          v8f acc = {0,0,0,0,0,0,0,0};
#pragma unroll
          for (int kk = 0; kk < 4; ++kk) {
            v16h a = load_a(bufY, 128, mt * 16, kk * 32, lane);
            v16h b = load_bm(fc1T, 128, c * 128 + nt * 16, kk * 32, lane);
            acc = wmma_f16(a, b, acc);
          }
          int hi = (lane >> 4) & 1, n = nt * 16 + (lane & 15);
          float bias = kp.fc1_b[l][c * 128 + n];
#pragma unroll
          for (int r = 0; r < 8; ++r) {
            float t = acc[r] + bias;
            float ge = 0.5f * t * (1.f + erff(t * 0.70710678118654752f));
            bufQ[(mt * 16 + hi * 8 + r) * 128 + n] = (_Float16)ge;
          }
        }
        __syncthreads();
        // partial: acc2 += z_c @ fc2[c*128 : c*128+128, :]
#pragma unroll
        for (int nt = 0; nt < 8; ++nt) {
#pragma unroll
          for (int kk = 0; kk < 4; ++kk) {
            v16h a = load_a(bufQ, 128, wave * 16, kk * 32, lane);
            v16h b = load_bm(fc2T, 512, nt * 16, c * 128 + kk * 32, lane);
            acc2[nt] = wmma_f16(a, b, acc2[nt]);
          }
        }
        __syncthreads();
      }
      int hi = (lane >> 4) & 1;
#pragma unroll
      for (int nt = 0; nt < 8; ++nt) {
        int n = nt * 16 + (lane & 15);
        float bias = kp.fc2_b[l][n];
#pragma unroll
        for (int r = 0; r < 8; ++r) {
          int m = wave * 16 + hi * 8 + r;
          h[m * 128 + n] += acc2[nt][r] + bias;
        }
      }
    }
    __syncthreads();
  } // layers

  // ---- final LN (f32, into S) ----
  if (tid < 128) {
    const float* r = h + tid * 128;
    float m = 0.f;
    for (int j = 0; j < 128; ++j) m += r[j];
    m *= (1.f / 128.f);
    float v = 0.f;
    for (int j = 0; j < 128; ++j) { float d = r[j] - m; v += d * d; }
    float rs = rsqrtf(v * (1.f / 128.f) + 1e-5f);
    for (int j = 0; j < 128; ++j)
      S[tid * 128 + j] = (r[j] - m) * rs * kp.norm_g[j] + kp.norm_b[j];
  }
  __syncthreads();

  // ---- head: out[b, o, t, p] = yn[p] . head_w[:, o] + head_b[o] ----
  for (int idx = tid; idx < 128 * 33; idx += 256) {
    int p = idx / 33, o = idx - p * 33;
    const float* yr = S + p * 128;
    float acc = kp.head_b[o];
    for (int j = 0; j < 128; ++j) acc += yr[j] * kp.head_w[j * 33 + o];
    kp.out[(((size_t)(bb * 33 + o) * 256 + tt) << 7) + p] = acc;
  }
}

extern "C" void kernel_launch(void* const* d_in, const int* in_sizes, int n_in,
                              void* d_out, int out_size, void* d_ws, size_t ws_size,
                              hipStream_t stream) {
  (void)in_sizes; (void)n_in; (void)out_size; (void)ws_size;
  auto F = [&](int i) { return (const float*)d_in[i]; };
  _Float16* ws = (_Float16*)d_ws;

  // weight conversion/transpose to f16 (cached in L2 for all 1024 blocks)
  for (int l = 0; l < 2; ++l) {
    int base = 7 + 16 * l;
    w_to_f16_t<<<(128 * 384 + 255) / 256, 256, 0, stream>>>(F(base + 2),  ws + (size_t)l * LWS,          128, 384);
    w_to_f16_t<<<(128 * 128 + 255) / 256, 256, 0, stream>>>(F(base + 8),  ws + (size_t)l * LWS + 49152,  128, 128);
    w_to_f16_t<<<(128 * 512 + 255) / 256, 256, 0, stream>>>(F(base + 12), ws + (size_t)l * LWS + 65536,  128, 512);
    w_to_f16_t<<<(512 * 128 + 255) / 256, 256, 0, stream>>>(F(base + 14), ws + (size_t)l * LWS + 131072, 512, 128);
  }

  KParams kp;
  kp.x = F(0); kp.target = F(1);
  kp.cond_w = F(2); kp.cond_b = F(3);
  kp.inp_w = F(4);  kp.inp_b = F(5);
  kp.sos = F(6);
  for (int l = 0; l < 2; ++l) {
    int base = 7 + 16 * l;
    kp.ln1_g[l] = F(base + 0);  kp.ln1_b[l] = F(base + 1);
    kp.qkv_b[l] = F(base + 3);
    kp.qn_g[l]  = F(base + 4);  kp.qn_b[l]  = F(base + 5);
    kp.kn_g[l]  = F(base + 6);  kp.kn_b[l]  = F(base + 7);
    kp.proj_b[l] = F(base + 9);
    kp.ln2_g[l] = F(base + 10); kp.ln2_b[l] = F(base + 11);
    kp.fc1_b[l] = F(base + 13); kp.fc2_b[l] = F(base + 15);
  }
  kp.norm_g = F(39); kp.norm_b = F(40);
  kp.head_w = F(41); kp.head_b = F(42);
  kp.wh = ws;
  kp.out = (float*)d_out;

  hipFuncSetAttribute((const void*)pitchformer_main,
                      hipFuncAttributeMaxDynamicSharedMemorySize, SMEM_BYTES);
  pitchformer_main<<<1024, 256, SMEM_BYTES, stream>>>(kp);
}
